// BuNN_10797547782304
// MI455X (gfx1250) — matrix operand order repaired and checked by reading, not compiled, not run on hardware
//
#include <hip/hip_runtime.h>
#include <hip/hip_bf16.h>

typedef __attribute__((ext_vector_type(16))) _Float16 v16h;
typedef __attribute__((ext_vector_type(8)))  float    v8f;

static constexpr int NN   = 100000;   // nodes
static constexpr int EEDG = 1600000;  // edges
static constexpr int DD   = 128;      // total dim
static constexpr int BB   = 64;       // bundles
static constexpr int LAY  = 4;
static constexpr int KORD = 8;
static constexpr int DOUT = 40;

__device__ __forceinline__ float gelu_exact(float v) {
    return 0.5f * v * (1.0f + erff(v * 0.70710678118654752f));
}

// -------------------- WMMA GEMM: C = A(f32, n x 128) * Bt(f16, 128 x ldB) + bias --------------------
// One wave computes one 16x16 output tile. K = 128 fixed (4 steps of 32).
template <int GELU>
__global__ void gemm_wmma_kernel(const float* __restrict__ A,
                                 const _Float16* __restrict__ Bt,
                                 const float* __restrict__ bias,
                                 float* __restrict__ C,
                                 int nrows, int dout, int ldB, int ldC) {
    const int lane  = threadIdx.x & 31;
    const int wave  = threadIdx.x >> 5;
    const int tn_ct = (dout + 15) >> 4;
    const int tm_ct = nrows >> 4;
    const int tile  = blockIdx.x * (blockDim.x >> 5) + wave;
    if (tile >= tm_ct * tn_ct) return;              // uniform per wave
    const int tm = tile / tn_ct;
    const int tn = tile % tn_ct;
    const int hsel = lane >> 4;    // 0: lanes 0-15, 1: lanes 16-31
    const int l15  = lane & 15;

    const float* arow = A + (size_t)(tm * 16 + l15) * DD;
    v8f acc = {};
#pragma unroll
    for (int k0 = 0; k0 < DD; k0 += 32) {
        // A fragment: 16-bit A 16x32 layout (lanes 0-15: K 0-7 & 16-23; lanes 16-31: K 8-15 & 24-31)
        v16h a;
        const float* p0 = arow + k0 + hsel * 8;
#pragma unroll
        for (int i = 0; i < 8; ++i) {
            a[i]     = (_Float16)p0[i];
            a[i + 8] = (_Float16)p0[16 + i];
        }
        // B fragment: lane l holds B[K][N=tn*16+l%16], K in same order as A (lanes 0-15: K 0-7 & 16-23; lanes 16-31: K 8-15 & 24-31)
        const _Float16* bp = Bt + (size_t)(k0 + hsel * 8) * ldB + tn * 16 + l15;
        v16h b;
#pragma unroll
        for (int i = 0; i < 8; ++i) {
            b[i]     = bp[(size_t)i * ldB];
            b[i + 8] = bp[(size_t)(16 + i) * ldB];
        }
        acc = __builtin_amdgcn_wmma_f32_16x16x32_f16(false, a, false, b,
                                                     (short)0, acc, false, false);
    }
    // C/D layout: lanes 0-15 -> N=lane, M=v; lanes 16-31 -> N=lane-16, M=8+v
    const int col   = tn * 16 + l15;
    const bool cok  = col < dout;
    const float bv  = cok ? bias[col] : 0.0f;
#pragma unroll
    for (int v = 0; v < 8; ++v) {
        const int row = tm * 16 + hsel * 8 + v;
        float val = acc[v] + bv;
        if (GELU) val = gelu_exact(val);
        if (cok) C[(size_t)row * ldC + col] = val;
    }
}

// -------------------- weight transpose + f16 convert: W(dout x 128) -> Wt(128 x ldB) --------------------
__global__ void wtrans_f16_kernel(const float* __restrict__ W, _Float16* __restrict__ Wt,
                                  int dout, int ldB) {
    int i = blockIdx.x * blockDim.x + threadIdx.x;
    if (i >= DD * ldB) return;
    int k = i / ldB;
    int n = i % ldB;
    Wt[i] = (n < dout) ? (_Float16)W[(size_t)n * DD + k] : (_Float16)0.0f;
}

// -------------------- graph prep --------------------
__global__ void zero_kernel(float* __restrict__ p, size_t cnt) {
    size_t i = (size_t)blockIdx.x * blockDim.x + threadIdx.x;
    if (i < cnt) p[i] = 0.0f;
}

__global__ void degree_kernel(const int* __restrict__ src, const int* __restrict__ dst,
                              float* __restrict__ degO, float* __restrict__ degI, int e) {
    int i = blockIdx.x * blockDim.x + threadIdx.x;
    if (i >= e) return;
    atomicAdd(&degO[src[i]], 1.0f);
    atomicAdd(&degI[dst[i]], 1.0f);
}

__global__ void enorm_kernel(const int* __restrict__ src, const int* __restrict__ dst,
                             const float* __restrict__ degO, const float* __restrict__ degI,
                             float* __restrict__ en, int e) {
    int i = blockIdx.x * blockDim.x + threadIdx.x;
    if (i >= e) return;
    float a = fmaxf(degO[src[i]], 1.0f);
    float b = fmaxf(degI[dst[i]], 1.0f);
    en[i] = rsqrtf(a * b);
}

// -------------------- diffusion --------------------
// One wave per edge; each lane moves 4 contiguous floats (float4 gather, 4 atomic f32 adds).
__global__ void scatter_kernel(const int* __restrict__ src, const int* __restrict__ dst,
                               const float* __restrict__ en,
                               const float* __restrict__ term, float* __restrict__ agg, int e) {
    long long tid = (long long)blockIdx.x * blockDim.x + threadIdx.x;
    int edge = (int)(tid >> 5);
    int lane = (int)(tid & 31);
    if (edge >= e) return;
    const int s = src[edge];
    const int d = dst[edge];
    const float w = en[edge];
    const float4 tv = *(const float4*)(term + (size_t)s * DD + lane * 4);
    float* ap = agg + (size_t)d * DD + lane * 4;
    atomicAdd(&ap[0], w * tv.x);
    atomicAdd(&ap[1], w * tv.y);
    atomicAdd(&ap[2], w * tv.z);
    atomicAdd(&ap[3], w * tv.w);
}

// term = coef * (term - agg); result += term
__global__ void diffuse_update_kernel(float* __restrict__ term, const float* __restrict__ agg,
                                      float* __restrict__ result, float coef, size_t cnt) {
    size_t i = (size_t)blockIdx.x * blockDim.x + threadIdx.x;
    if (i >= cnt) return;
    float t = coef * (term[i] - agg[i]);
    term[i] = t;
    result[i] += t;
}

// -------------------- rotations --------------------
// transpose=True: xr = c*x - s*y ; yr = s*x + c*y
__global__ void rotate_t_kernel(const float* __restrict__ h, const float* __restrict__ ang,
                                float* __restrict__ out, int n) {
    int i = blockIdx.x * blockDim.x + threadIdx.x;
    if (i >= n * BB) return;
    int node = i >> 6;
    int b = i & (BB - 1);
    float a = ang[(size_t)node * BB + b];
    float c, s;
    __sincosf(a, &s, &c);
    const float* hp = h + (size_t)node * DD + 2 * b;
    float x = hp[0], y = hp[1];
    float* op = out + (size_t)node * DD + 2 * b;
    op[0] = c * x - s * y;
    op[1] = s * x + c * y;
}

// transpose=False fused with residual: h += gelu(rotate(Hd))
__global__ void rotate_gelu_add_kernel(const float* __restrict__ Hd, const float* __restrict__ ang,
                                       float* __restrict__ h, int n) {
    int i = blockIdx.x * blockDim.x + threadIdx.x;
    if (i >= n * BB) return;
    int node = i >> 6;
    int b = i & (BB - 1);
    float a = ang[(size_t)node * BB + b];
    float c, s;
    __sincosf(a, &s, &c);
    const float* hp = Hd + (size_t)node * DD + 2 * b;
    float x = hp[0], y = hp[1];
    float xr = c * x + s * y;
    float yr = -s * x + c * y;
    float* op = h + (size_t)node * DD + 2 * b;
    op[0] += gelu_exact(xr);
    op[1] += gelu_exact(yr);
}

// -------------------- orchestration --------------------
extern "C" void kernel_launch(void* const* d_in, const int* in_sizes, int n_in,
                              void* d_out, int out_size, void* d_ws, size_t ws_size,
                              hipStream_t stream) {
    const float* x      = (const float*)d_in[0];
    const int*   ei     = (const int*)d_in[1];
    const int*   src    = ei;
    const int*   dst    = ei + EEDG;
    const float* in_w   = (const float*)d_in[2];
    const float* in_b   = (const float*)d_in[3];
    const float* phi_w1 = (const float*)d_in[4];
    const float* phi_b1 = (const float*)d_in[5];
    const float* phi_w2 = (const float*)d_in[6];
    const float* phi_b2 = (const float*)d_in[7];
    const float* lt_w   = (const float*)d_in[8];
    const float* lt_b   = (const float*)d_in[9];
    const float* out_w  = (const float*)d_in[10];
    const float* out_b  = (const float*)d_in[11];
    float* out = (float*)d_out;

    // workspace carve-out (256B aligned)
    char* wsp = (char*)d_ws;
    auto carve = [&](size_t bytes) -> char* {
        char* p = wsp;
        wsp += (bytes + 255) & ~(size_t)255;
        return p;
    };
    const size_t nodeElems = (size_t)NN * DD;
    float*    enorm = (float*)carve((size_t)EEDG * 4);
    float*    degO  = (float*)carve((size_t)NN * 4);
    float*    degI  = (float*)carve((size_t)NN * 4);
    float*    h     = (float*)carve(nodeElems * 4);
    float*    bufA  = (float*)carve(nodeElems * 4);   // gelu-mlp temp / rotated input
    float*    bufB  = (float*)carve(nodeElems * 4);   // H, then diffusion result
    float*    bufC  = (float*)carve(nodeElems * 4);   // term
    float*    bufD  = (float*)carve(nodeElems * 4);   // agg
    float*    ang   = (float*)carve((size_t)NN * BB * 4);
    _Float16* WtIn  = (_Float16*)carve((size_t)DD * DD * 2);
    _Float16* WtP1  = (_Float16*)carve((size_t)LAY * DD * DD * 2);
    _Float16* WtP2  = (_Float16*)carve((size_t)LAY * DD * BB * 2);
    _Float16* WtLt  = (_Float16*)carve((size_t)LAY * DD * DD * 2);
    _Float16* WtOut = (_Float16*)carve((size_t)DD * 48 * 2);

    const int TPB = 256;
    auto blocks1d = [&](size_t work) { return (int)((work + TPB - 1) / TPB); };

    // ---- degrees / edge norms ----
    zero_kernel<<<blocks1d(NN), TPB, 0, stream>>>(degO, NN);
    zero_kernel<<<blocks1d(NN), TPB, 0, stream>>>(degI, NN);
    degree_kernel<<<blocks1d(EEDG), TPB, 0, stream>>>(src, dst, degO, degI, EEDG);
    enorm_kernel<<<blocks1d(EEDG), TPB, 0, stream>>>(src, dst, degO, degI, enorm, EEDG);

    // ---- weight transpose + f16 convert ----
    auto wtrans = [&](const float* W, _Float16* Wt, int dout, int ldB) {
        wtrans_f16_kernel<<<blocks1d((size_t)DD * ldB), TPB, 0, stream>>>(W, Wt, dout, ldB);
    };
    wtrans(in_w, WtIn, DD, DD);
    for (int l = 0; l < LAY; ++l) {
        wtrans(phi_w1 + (size_t)l * DD * DD, WtP1 + (size_t)l * DD * DD, DD, DD);
        wtrans(phi_w2 + (size_t)l * BB * DD, WtP2 + (size_t)l * DD * BB, BB, BB);
        wtrans(lt_w   + (size_t)l * DD * DD, WtLt + (size_t)l * DD * DD, DD, DD);
    }
    wtrans(out_w, WtOut, DOUT, 48);

    auto gemm = [&](const float* A, const _Float16* Bt, const float* bias, float* C,
                    int dout, int ldB, int ldC, bool gelu) {
        int tiles = (NN / 16) * ((dout + 15) / 16);
        int blk = (tiles + 7) / 8;   // 8 waves / 256-thread block
        if (gelu)
            gemm_wmma_kernel<1><<<blk, TPB, 0, stream>>>(A, Bt, bias, C, NN, dout, ldB, ldC);
        else
            gemm_wmma_kernel<0><<<blk, TPB, 0, stream>>>(A, Bt, bias, C, NN, dout, ldB, ldC);
    };

    // ---- input projection: h = x @ in_w^T + in_b ----
    gemm(x, WtIn, in_b, h, DD, DD, DD, false);

    // ---- layers ----
    for (int l = 0; l < LAY; ++l) {
        // phi MLP: bufA = gelu(h @ W1^T + b1) ; ang = bufA @ W2^T + b2
        gemm(h, WtP1 + (size_t)l * DD * DD, phi_b1 + (size_t)l * DD, bufA, DD, DD, DD, true);
        gemm(bufA, WtP2 + (size_t)l * DD * BB, phi_b2 + (size_t)l * BB, ang, BB, BB, BB, false);
        // bufA = rotate_T(h, ang)
        rotate_t_kernel<<<blocks1d((size_t)NN * BB), TPB, 0, stream>>>(h, ang, bufA, NN);
        // bufB = bufA @ lt_w^T + lt_b
        gemm(bufA, WtLt + (size_t)l * DD * DD, lt_b + (size_t)l * DD, bufB, DD, DD, DD, false);
        // diffusion: result(bufB) = H; term(bufC) = H; 8 Taylor steps
        hipMemcpyAsync(bufC, bufB, nodeElems * 4, hipMemcpyDeviceToDevice, stream);
        for (int k = 1; k <= KORD; ++k) {
            zero_kernel<<<blocks1d(nodeElems), TPB, 0, stream>>>(bufD, nodeElems);
            scatter_kernel<<<blocks1d((size_t)EEDG * 32), TPB, 0, stream>>>(
                src, dst, enorm, bufC, bufD, EEDG);
            diffuse_update_kernel<<<blocks1d(nodeElems), TPB, 0, stream>>>(
                bufC, bufD, bufB, -1.0f / (float)k, nodeElems);
        }
        // h += gelu(rotate(bufB, ang, transpose=false))
        rotate_gelu_add_kernel<<<blocks1d((size_t)NN * BB), TPB, 0, stream>>>(bufB, ang, h, NN);
    }

    // ---- output projection ----
    gemm(h, WtOut, out_b, out, DOUT, 48, DOUT, false);
}
